// MAB_43885975830884
// MI455X (gfx1250) — matrix operand order repaired and checked by reading, not compiled, not run on hardware
//
#include <hip/hip_runtime.h>

typedef __attribute__((ext_vector_type(16))) _Float16 v16h;
typedef __attribute__((ext_vector_type(4)))  _Float16 h4;
typedef __attribute__((ext_vector_type(8)))  float    v8f;
typedef __attribute__((ext_vector_type(4)))  unsigned int u32x4;
typedef __attribute__((ext_vector_type(8)))  int      i32x8;
typedef __attribute__((ext_vector_type(4)))  int      i32x4;

// ---------------------------------------------------------------------------
// CDNA5 WMMA helpers (wave32).  D = A(16x32 f16) * B(32x16 f16) + C(16x16 f32)
// ---------------------------------------------------------------------------
static __device__ inline v8f wmma16(v16h a, v16h b, v8f c) {
  return __builtin_amdgcn_wmma_f32_16x16x32_f16(
      /*neg_a=*/false, a, /*neg_b=*/false, b,
      /*c_mod=*/(short)0, c, /*reuse_a=*/false, /*reuse_b=*/false);
}

// A fragment (16x32, MxK) from row-major LDS tile As[ld] at origin (m0,k0).
// ISA 7.12.2: lane L holds row m=L&15; VGPR i halves h map to
// k = (i>>2)*16 + (L>=16)*8 + (i&3)*2 + h.
static __device__ inline v16h load_a(const _Float16* As, int ld, int m0, int k0,
                                     int lane) {
  const int m  = m0 + (lane & 15);
  const int hi = (lane >> 4) & 1;
  v16h a;
#pragma unroll
  for (int i = 0; i < 8; ++i) {
    const int k = k0 + ((i >> 2) << 4) + (hi << 3) + ((i & 3) << 1);
    a[2 * i]     = As[m * ld + k];
    a[2 * i + 1] = As[m * ld + k + 1];
  }
  return a;
}

// B fragment (32x16, KxN) from LDS tile stored [n][k] (ld halves) at (n0,k0).
// Lane L holds column n=L&15; VGPR j halves h map to
// k = (j>>2)*16 + (L>=16)*8 + (j&3)*2 + h.
static __device__ inline v16h load_b(const _Float16* Bs, int ld, int n0, int k0,
                                     int lane) {
  const int n  = n0 + (lane & 15);
  const int hi = (lane >> 4) & 1;
  v16h b;
#pragma unroll
  for (int j = 0; j < 8; ++j) {
    const int k = k0 + ((j >> 2) << 4) + (hi << 3) + ((j & 3) << 1);
    b[2 * j]     = Bs[n * ld + k];
    b[2 * j + 1] = Bs[n * ld + k + 1];
  }
  return b;
}

// ---------------------------------------------------------------------------
// Tensor Data Mover: 2D f16 tile (rows x cols) global -> LDS, with LDS
// padding of 1 DWORD after every 32 DWORDs (128B row) so a 64-wide f16 row
// lands on a 66-half pitch (bank-conflict-free fragment reads).
// D# layout per CDNA5 ISA ch.8; 6-arg builtin (this toolchain's signature).
// ---------------------------------------------------------------------------
static __device__ inline void tdm_load_tile_f16(unsigned lds_off,
                                                const _Float16* gptr,
                                                unsigned rows, unsigned cols,
                                                unsigned row_stride_elems) {
  const unsigned long long ga = (unsigned long long)(size_t)gptr;
  u32x4 g0;
  g0[0] = 1u;                                   // count=1 (valid user D#)
  g0[1] = lds_off;                              // lds_addr (bytes)
  g0[2] = (unsigned)(ga & 0xFFFFFFFFu);         // global_addr[31:0]
  g0[3] = (unsigned)((ga >> 32) & 0x1FFFFFFu)   // global_addr[56:32]
          | (2u << 30);                         // type = 2 ("image")
  i32x8 g1;
  // data_size=1 (2B) | pad_enable | pad_interval=4 (32 DW) | pad_amount=0 (1 DW)
  g1[0] = (int)((1u << 16) | (1u << 20) | (4u << 22));
  g1[1] = (int)((cols & 0xFFFFu) << 16);                        // tensor_dim0 lo
  g1[2] = (int)(((cols >> 16) & 0xFFFFu) | ((rows & 0xFFFFu) << 16)); // td0 hi | td1 lo
  g1[3] = (int)(((rows >> 16) & 0xFFFFu) | ((cols & 0xFFFFu) << 16)); // td1 hi | tile_dim0
  g1[4] = (int)(rows & 0xFFFFu);                                // tile_dim1 (tile_dim2=0)
  g1[5] = (int)row_stride_elems;                                // tensor_dim0_stride lo
  g1[6] = 0;
  g1[7] = 0;
  const i32x4 z4 = {0, 0, 0, 0};                // groups 2/3 unused (2D tensor)
  const i32x8 z8 = {0, 0, 0, 0, 0, 0, 0, 0};    // extra group (zero-filled)
  __builtin_amdgcn_tensor_load_to_lds(g0, g1, z4, z4, z8, 0);
}

// ---------------------------------------------------------------------------
// GEMM: Y[m,n] = X[m,:] . W[n,:] + bias[n]        (torch Linear, N = K = 256)
// MODE 0: write Yf32 and Yf16;  MODE 1: write Yf16 only;
// MODE 2: Y = resid + relu(Y), write Yf32;
// MODE 3: write Yf16 in per-head transposed layout [B,H,64,seq] (for V).
// Block: 128 thr (4 waves), 64x64 output tile, K-steps of 32 through LDS.
// ---------------------------------------------------------------------------
template <int MODE>
__global__ __launch_bounds__(128) void gemm_xwT_kernel(
    const float* __restrict__ X, const float* __restrict__ W,
    const float* __restrict__ bias, float* __restrict__ Yf32,
    _Float16* __restrict__ Yf16, const float* __restrict__ resid) {
  __shared__ _Float16 Xs[64][36];  // [m][k], 72B rows (8B aligned)
  __shared__ _Float16 Ws[64][36];  // [n][k]
  const int t = threadIdx.x, lane = t & 31, w = t >> 5;
  const int m0 = blockIdx.x * 64, n0 = blockIdx.y * 64;
  const int mw = (w & 1) * 32, nw = (w >> 1) * 32;
  v8f acc[2][2] = {};

  for (int k0 = 0; k0 < 256; k0 += 32) {
    __syncthreads();
#pragma unroll
    for (int e = 0; e < 4; ++e) {
      const int idx = t + e * 128;            // 0..511 float4 units
      const int r = idx >> 3, c4 = idx & 7;   // row, float4 col
      const float4 xv =
          ((const float4*)(X + (size_t)(m0 + r) * 256 + k0))[c4];
      const float4 wv =
          ((const float4*)(W + (size_t)(n0 + r) * 256 + k0))[c4];
      *(h4*)&Xs[r][c4 * 4] =
          (h4){(_Float16)xv.x, (_Float16)xv.y, (_Float16)xv.z, (_Float16)xv.w};
      *(h4*)&Ws[r][c4 * 4] =
          (h4){(_Float16)wv.x, (_Float16)wv.y, (_Float16)wv.z, (_Float16)wv.w};
    }
    __syncthreads();
    const v16h a0 = load_a(&Xs[0][0], 36, mw, 0, lane);
    const v16h a1 = load_a(&Xs[0][0], 36, mw + 16, 0, lane);
    const v16h b0 = load_b(&Ws[0][0], 36, nw, 0, lane);
    const v16h b1 = load_b(&Ws[0][0], 36, nw + 16, 0, lane);
    acc[0][0] = wmma16(a0, b0, acc[0][0]);
    acc[0][1] = wmma16(a0, b1, acc[0][1]);
    acc[1][0] = wmma16(a1, b0, acc[1][0]);
    acc[1][1] = wmma16(a1, b1, acc[1][1]);
  }

  const int hi = lane >> 4, nl = lane & 15;
#pragma unroll
  for (int ti = 0; ti < 2; ++ti)
#pragma unroll
    for (int tj = 0; tj < 2; ++tj)
#pragma unroll
      for (int r = 0; r < 8; ++r) {
        const size_t m = m0 + mw + ti * 16 + r + hi * 8;
        const size_t n = n0 + nw + tj * 16 + nl;
        float y = acc[ti][tj][r] + bias[n];
        if (MODE == 2) {
          y = resid[m * 256 + n] + fmaxf(y, 0.0f);
          Yf32[m * 256 + n] = y;
        } else if (MODE == 0) {
          Yf32[m * 256 + n] = y;
          Yf16[m * 256 + n] = (_Float16)y;
        } else if (MODE == 1) {
          Yf16[m * 256 + n] = (_Float16)y;
        } else {  // MODE 3: vT[b][head][d][seq]
          const size_t bi = m >> 11, seq = m & 2047;
          const size_t head = n >> 6, dl = n & 63;
          Yf16[((bi * 4 + head) * 64 + dl) * 2048 + seq] = (_Float16)y;
        }
      }
}

// ---------------------------------------------------------------------------
// Flash attention per (b, h, 64-row q block).  head dim = 64, Nk = 2048.
// scores scaled by 1/sqrt(D)=1/16; O = q_f32 + softmax(S) @ V (per-head
// residual).  K tiles arrive row-major [kv][d]; V tiles arrive pre-transposed
// [d][kv] (from the MODE-3 projection), both via double-buffered TDM loads
// issued by wave 0 and synchronized with TENSORcnt.  Block: 128 thr (4
// waves); wave w owns q rows [16w,16w+16) and all 64 head columns.
// ---------------------------------------------------------------------------
__global__ __launch_bounds__(128) void attn_kernel(
    const _Float16* __restrict__ qh, const _Float16* __restrict__ kh,
    const _Float16* __restrict__ vT, const float* __restrict__ qf,
    float* __restrict__ out) {
  __shared__ _Float16 Qs[64][66];      // [m][d]
  __shared__ _Float16 Ks[2][64][66];   // [kv][d]  (B layout for S = Q.K^T)
  __shared__ _Float16 Vs[2][64][66];   // [d][kv]  (B layout for O = P.V)
  __shared__ _Float16 Ps[64][66];      // [m][kv]  (A layout for O = P.V)
  __shared__ float rowmax[64], rowsum[64], rowscale[64];

  const int t = threadIdx.x, lane = t & 31, w = t >> 5;
  const int hi = lane >> 4, nl = lane & 15;
  const int q0 = blockIdx.x * 64;
  const int h  = blockIdx.y;
  const int b  = blockIdx.z;
  const size_t baseQ = ((size_t)b * 2048 + q0) * 256 + h * 64;
  const size_t baseK = (size_t)b * 2048 * 256 + h * 64;
  const size_t baseV = (size_t)(b * 4 + h) * 64 * 2048;  // vT[b][h][0][0]
  const float sc = 0.0625f;  // 1/sqrt(256)
  const int NT = 2048 / 64;  // 32 k-tiles

  // Q tile via TDM (issued once, wave 0).
  if (w == 0) {
    tdm_load_tile_f16((unsigned)(size_t)&Qs[0][0], qh + baseQ, 64, 64, 256);
    __builtin_amdgcn_s_wait_tensorcnt(0);
  }
  if (t < 64) { rowmax[t] = -1e30f; rowsum[t] = 0.0f; }
  __syncthreads();

  // Q fragments are loop-invariant.
  const v16h qa0 = load_a(&Qs[0][0], 66, w * 16, 0, lane);
  const v16h qa1 = load_a(&Qs[0][0], 66, w * 16, 32, lane);
  v8f acco[4] = {};

  // Prefetch k-tile 0.
  if (w == 0) {
    tdm_load_tile_f16((unsigned)(size_t)&Ks[0][0][0], kh + baseK, 64, 64, 256);
    tdm_load_tile_f16((unsigned)(size_t)&Vs[0][0][0], vT + baseV, 64, 64, 2048);
  }

  for (int kt = 0; kt < NT; ++kt) {
    __syncthreads();  // all waves done with buffer (kt+1)&1 from iter kt-1
    if (w == 0) {
      if (kt + 1 < NT) {
        const int nb = (kt + 1) & 1;
        tdm_load_tile_f16((unsigned)(size_t)&Ks[nb][0][0],
                          kh + baseK + (size_t)(kt + 1) * 64 * 256, 64, 64, 256);
        tdm_load_tile_f16((unsigned)(size_t)&Vs[nb][0][0],
                          vT + baseV + (size_t)(kt + 1) * 64, 64, 64, 2048);
        __builtin_amdgcn_s_wait_tensorcnt(2);  // current tile done, prefetch in flight
      } else {
        __builtin_amdgcn_s_wait_tensorcnt(0);
      }
    }
    __syncthreads();
    const _Float16* Kc = &Ks[kt & 1][0][0];
    const _Float16* Vc = &Vs[kt & 1][0][0];

    // S tile: 16 q-rows x 64 kv  (4 WMMA n-tiles, 2 K-steps each)
    v8f s[4];
#pragma unroll
    for (int nt = 0; nt < 4; ++nt) {
      v8f c = {};
      c = wmma16(qa0, load_b(Kc, 66, nt * 16, 0, lane), c);
      c = wmma16(qa1, load_b(Kc, 66, nt * 16, 32, lane), c);
      s[nt] = c;
    }

    // online softmax: row max (16-lane groups share a row per C-layout)
#pragma unroll
    for (int r = 0; r < 8; ++r) {
      float mx = -1e30f;
#pragma unroll
      for (int nt = 0; nt < 4; ++nt) mx = fmaxf(mx, s[nt][r] * sc);
#pragma unroll
      for (int off = 1; off < 16; off <<= 1)
        mx = fmaxf(mx, __shfl_xor(mx, off, 32));
      const int row = w * 16 + r + hi * 8;
      if (nl == 0) {
        const float mo = rowmax[row];
        const float mn = fmaxf(mo, mx);
        rowmax[row]   = mn;
        rowscale[row] = __expf(mo - mn);
      }
    }

    // p = exp(s - m_new); write A-layout tile; update l; rescale O
#pragma unroll
    for (int r = 0; r < 8; ++r) {
      const int row = w * 16 + r + hi * 8;
      const float mn = rowmax[row];
      float psum = 0.0f;
#pragma unroll
      for (int nt = 0; nt < 4; ++nt) {
        const float p = __expf(s[nt][r] * sc - mn);
        psum += p;
        Ps[row][nt * 16 + nl] = (_Float16)p;
      }
#pragma unroll
      for (int off = 1; off < 16; off <<= 1) psum += __shfl_xor(psum, off, 32);
      const float scale = rowscale[row];
      if (nl == 0) rowsum[row] = rowsum[row] * scale + psum;
#pragma unroll
      for (int dt = 0; dt < 4; ++dt) acco[dt][r] *= scale;
    }

    // O += P(16x64) @ V(64x64)   (V tile already [d][kv] -> fast B reads)
    const v16h pa0 = load_a(&Ps[0][0], 66, w * 16, 0, lane);
    const v16h pa1 = load_a(&Ps[0][0], 66, w * 16, 32, lane);
#pragma unroll
    for (int dt = 0; dt < 4; ++dt) {
      acco[dt] = wmma16(pa0, load_b(Vc, 66, dt * 16, 0, lane), acco[dt]);
      acco[dt] = wmma16(pa1, load_b(Vc, 66, dt * 16, 32, lane), acco[dt]);
    }
  }

  // epilogue: out = q_residual + O / l
#pragma unroll
  for (int dt = 0; dt < 4; ++dt)
#pragma unroll
    for (int r = 0; r < 8; ++r) {
      const int row = w * 16 + r + hi * 8;
      const int dcol = dt * 16 + nl;
      const size_t gi = ((size_t)b * 2048 + q0 + row) * 256 + h * 64 + dcol;
      out[gi] = qf[gi] + acco[dt][r] / rowsum[row];
    }
}

// ---------------------------------------------------------------------------
// LayerNorm over rows of 256.  One wave per row, 8 rows per 256-thr block.
// (x may alias y: all reads complete before writes within the owning wave.)
// ---------------------------------------------------------------------------
__global__ __launch_bounds__(256) void ln_kernel(const float* x,
                                                 const float* __restrict__ g,
                                                 const float* __restrict__ bb,
                                                 float* y) {
  const int lane = threadIdx.x & 31, w = threadIdx.x >> 5;
  const size_t row = (size_t)blockIdx.x * 8 + w;
  const float* xr = x + row * 256;
  float v[8], s = 0.0f;
#pragma unroll
  for (int j = 0; j < 8; ++j) { v[j] = xr[lane + j * 32]; s += v[j]; }
#pragma unroll
  for (int off = 1; off < 32; off <<= 1) s += __shfl_xor(s, off, 32);
  const float mu = s * (1.0f / 256.0f);
  float q = 0.0f;
#pragma unroll
  for (int j = 0; j < 8; ++j) { const float d = v[j] - mu; q += d * d; }
#pragma unroll
  for (int off = 1; off < 32; off <<= 1) q += __shfl_xor(q, off, 32);
  const float inv = rsqrtf(q * (1.0f / 256.0f) + 1e-5f);
#pragma unroll
  for (int j = 0; j < 8; ++j) {
    const int c = lane + j * 32;
    y[row * 256 + c] = (v[j] - mu) * inv * g[c] + bb[c];
  }
}

// ---------------------------------------------------------------------------
extern "C" void kernel_launch(void* const* d_in, const int* in_sizes, int n_in,
                              void* d_out, int out_size, void* d_ws,
                              size_t ws_size, hipStream_t stream) {
  (void)in_sizes; (void)n_in; (void)out_size; (void)ws_size;
  const float* Q  = (const float*)d_in[0];
  const float* K  = (const float*)d_in[1];
  const float* Wq = (const float*)d_in[2];
  const float* bq = (const float*)d_in[3];
  const float* Wk = (const float*)d_in[4];
  const float* bk = (const float*)d_in[5];
  const float* Wv = (const float*)d_in[6];
  const float* bv = (const float*)d_in[7];
  const float* Wo = (const float*)d_in[8];
  const float* bo = (const float*)d_in[9];
  const float* g0 = (const float*)d_in[10];
  const float* b0 = (const float*)d_in[11];
  const float* g1 = (const float*)d_in[12];
  const float* b1 = (const float*)d_in[13];

  // M = B*N = 16384 rows, D = 256.
  char* ws = (char*)d_ws;
  float*    q_f32 = (float*)(ws + 0);            // 16.8 MB (reused as MLP out)
  _Float16* q_h   = (_Float16*)(ws + 16777216);  // 8.4 MB
  _Float16* k_h   = (_Float16*)(ws + 25165824);  // 8.4 MB
  _Float16* v_t   = (_Float16*)(ws + 33554432);  // 8.4 MB  [B,H,64,2048]
  float*    att   = (float*)(ws + 41943040);     // 16.8 MB (attn out / LN1 in place)
  float*    mlp   = q_f32;                       // alias: q_f32 dead after attn

  const dim3 gGemm(256, 4), bGemm(128);
  // q/k/v projections (f16 for WMMA; q also in f32 for the per-head residual;
  // v written pre-transposed per head)
  gemm_xwT_kernel<0><<<gGemm, bGemm, 0, stream>>>(Q, Wq, bq, q_f32, q_h, nullptr);
  gemm_xwT_kernel<1><<<gGemm, bGemm, 0, stream>>>(K, Wk, bk, nullptr, k_h, nullptr);
  gemm_xwT_kernel<3><<<gGemm, bGemm, 0, stream>>>(K, Wv, bv, nullptr, v_t, nullptr);

  // flash attention + per-head residual (TDM-fed tiles)
  attn_kernel<<<dim3(32, 4, 8), dim3(128), 0, stream>>>(q_h, k_h, v_t, q_f32, att);

  // LN1 (in place)
  ln_kernel<<<dim3(2048), dim3(256), 0, stream>>>(att, g0, b0, att);

  // O = O + relu(O @ Wo^T + bo)   (fused epilogue)
  gemm_xwT_kernel<2><<<gGemm, bGemm, 0, stream>>>(att, Wo, bo, mlp, nullptr, att);

  // LN2 -> d_out
  ln_kernel<<<dim3(2048), dim3(256), 0, stream>>>(mlp, g1, b1, (float*)d_out);
}
